// ProtoClVerbalizer_22969485099283
// MI455X (gfx1250) — matrix operand order repaired and checked by reading, not compiled, not run on hardware
//
#include <hip/hip_runtime.h>
#include <hip/hip_bf16.h>
#include <math.h>

// Problem constants (from reference)
#define NCLS   64
#define NUM    128
#define IN_DIM 768
#define HID    128
#define INV_SCALE 20.0f

typedef _Float16 h4  __attribute__((ext_vector_type(4)));
typedef _Float16 h8  __attribute__((ext_vector_type(8)));
typedef _Float16 v16h __attribute__((ext_vector_type(16)));
typedef float    v8f __attribute__((ext_vector_type(8)));
typedef float    f4  __attribute__((ext_vector_type(4)));
typedef int      v4i __attribute__((ext_vector_type(4)));

#define SHUF16(lo, hi) __builtin_shufflevector((lo), (hi), 0,1,2,3,4,5,6,7,8,9,10,11,12,13,14,15)

// Workspace layout (bytes, all offsets 256B aligned). Total ~10.75 MB.
#define OFF_NXH     ((size_t)0)          // 8192x128 f16 row-major      = 2 MB
#define OFF_NXHT    ((size_t)2097152)    // 64 x [128k][128n] f16 transp= 2 MB
#define OFF_NPROTO  ((size_t)4194304)    // 64x128 f32                  = 32 KB
#define OFF_COLPART ((size_t)4227072)    // 64x8192 f32                 = 2 MB
#define OFF_COLSUM  ((size_t)6324224)    // 8192 f32                    = 32 KB
#define OFF_DIAG    ((size_t)6356992)    // 64x128x128 f32              = 4 MB
#define OFF_PL      ((size_t)10551296)   // 1024 f32 proto-loss partials
#define OFF_IL      ((size_t)10555392)   // 64 f32 instance-loss partials
#define OFF_WT      ((size_t)10555648)   // 768x128 f16 head_w^T        = 192 KB

// ---- CDNA5 async global->LDS copy (ASYNCcnt path), guarded ---------------
#if defined(__has_builtin)
#if __has_builtin(__builtin_amdgcn_global_load_async_to_lds_b128)
#define HAVE_ASYNC_LDS 1
#endif
#endif

#ifdef HAVE_ASYNC_LDS
typedef __attribute__((address_space(1))) v4i* gas_ptr;   // prints as "__device__"
typedef __attribute__((address_space(3))) v4i* las_ptr;
static __device__ __forceinline__ void async_copy16(const void* g, void* l) {
  __builtin_amdgcn_global_load_async_to_lds_b128((gas_ptr)(void*)g, (las_ptr)l, 0, 0);
}
#endif

static __device__ __forceinline__ void wait_async0() {
#if defined(__has_builtin) && __has_builtin(__builtin_amdgcn_s_wait_asynccnt)
  __builtin_amdgcn_s_wait_asynccnt(0);
#else
  asm volatile("s_wait_asynccnt 0" ::: "memory");
#endif
}

// ---------------------------------------------------------------------------
// Gram kernel (first in TU so the disasm snippet shows it): block (b, a)
// computes the 128x128 block G = exp(nx_a . nx_b^T * 20). Both LDS tiles are
// straight copies (A from row-major nxh, B from pre-transposed nxhT) staged
// via CDNA5 async global->LDS (ASYNCcnt) when available.
// Emits per-block column-sum partials and, when a==b, the diagonal block.
// ---------------------------------------------------------------------------
__global__ __launch_bounds__(256) void gram_kernel(
    const _Float16* __restrict__ nxh, const _Float16* __restrict__ nxhT,
    float* __restrict__ colPart, float* __restrict__ diag) {
  __shared__ __align__(16) _Float16 AB[2 * 128 * 128];  // 64KB
  _Float16* Ah = AB;                 // [128][128] rows of class a
  _Float16* Bt = AB + 128 * 128;     // [128][128] Bt[k][n] = nx[b*128+n][k]

  const int t = threadIdx.x, lane = t & 31, w = t >> 5;
  const int b = blockIdx.x, a = blockIdx.y;

#ifdef HAVE_ASYNC_LDS
#pragma unroll
  for (int j = 0; j < 8; ++j) {
    int o = (t + 256 * j) * 8;       // half offset, 16B per op per lane
    async_copy16(nxh  + (size_t)a * 16384 + o, Ah + o);
    async_copy16(nxhT + (size_t)b * 16384 + o, Bt + o);
  }
  wait_async0();
  __syncthreads();
#else
#pragma unroll
  for (int j = 0; j < 8; ++j) {
    int o = (t + 256 * j) * 8;
    *(h8*)(Ah + o) = *(const h8*)(nxh + (size_t)a * 16384 + o);
    *(h8*)(Bt + o) = *(const h8*)(nxhT + (size_t)b * 16384 + o);
  }
  __syncthreads();
#endif

  const int wr = (w >> 1) * 32, wc = (w & 1) * 64;
  v8f z;
#pragma unroll
  for (int e = 0; e < 8; ++e) z[e] = 0.0f;
  v8f acc[2][4];
#pragma unroll
  for (int i = 0; i < 2; ++i)
#pragma unroll
    for (int j = 0; j < 4; ++j) acc[i][j] = z;

#pragma unroll
  for (int k0 = 0; k0 < 128; k0 += 32) {
    v16h af[2], bf[4];
    const int kbA = k0 + (lane >> 4) * 8;
#pragma unroll
    for (int rt = 0; rt < 2; ++rt) {
      int m = wr + rt * 16 + (lane & 15);
      h8 lo = *(const h8*)(Ah + m * 128 + kbA);
      h8 hi = *(const h8*)(Ah + m * 128 + kbA + 16);
      af[rt] = SHUF16(lo, hi);
    }
    const int kbB = k0 + (lane >> 4) * 8;
#pragma unroll
    for (int ct = 0; ct < 4; ++ct) {
      int n = wc + ct * 16 + (lane & 15);
#pragma unroll
      for (int s = 0; s < 8; ++s) {
        bf[ct][s] = Bt[(kbB + s) * 128 + n];
        bf[ct][s + 8] = Bt[(kbB + 16 + s) * 128 + n];
      }
    }
#pragma unroll
    for (int rt = 0; rt < 2; ++rt)
#pragma unroll
      for (int ct = 0; ct < 4; ++ct)
        acc[rt][ct] = __builtin_amdgcn_wmma_f32_16x16x32_f16(
            false, af[rt], false, bf[ct], (short)0, acc[rt][ct], false, false);
  }
  __syncthreads();                   // all LDS reads done; can alias colred

  // exp epilogue in-place
#pragma unroll
  for (int rt = 0; rt < 2; ++rt)
#pragma unroll
    for (int ct = 0; ct < 4; ++ct)
#pragma unroll
      for (int v = 0; v < 8; ++v)
        acc[rt][ct][v] = expf(acc[rt][ct][v] * INV_SCALE);

  // Column sums: C layout lane->N, VGPR->M; cross-half via shfl_xor 16.
  float* colred = (float*)AB;        // [4][128] aliases Ah
#pragma unroll
  for (int ct = 0; ct < 4; ++ct) {
    float s = 0.0f;
#pragma unroll
    for (int rt = 0; rt < 2; ++rt)
#pragma unroll
      for (int v = 0; v < 8; ++v) s += acc[rt][ct][v];
    s += __shfl_xor(s, 16, 32);      // add the other 16 rows
    if (lane < 16) {
      int n = wc + ct * 16 + lane;
      colred[(w >> 1) * 128 + n] = s;  // unique (slot, n) per writer
    }
  }
  __syncthreads();
  if (t < 128) {
    float s = colred[t] + colred[128 + t] + colred[256 + t] + colred[384 + t];
    colPart[(size_t)a * 8192 + b * 128 + t] = s;
  }

  if (a == b) {                      // store diagonal (pos) block
#pragma unroll
    for (int rt = 0; rt < 2; ++rt)
#pragma unroll
      for (int ct = 0; ct < 4; ++ct) {
        int n = wc + ct * 16 + (lane & 15);
        int mbase = wr + rt * 16 + ((lane >> 4) << 3);
#pragma unroll
        for (int v = 0; v < 8; ++v)
          diag[(size_t)b * 16384 + (mbase + v) * 128 + n] = acc[rt][ct][v];
      }
  }
}

// ---------------------------------------------------------------------------
// One-off: pre-transpose head_w to f16 wT[k][h] so stage A's B tiles are
// straight contiguous copies.
// ---------------------------------------------------------------------------
__global__ __launch_bounds__(256) void prep_wt_kernel(
    const float* __restrict__ head_w, _Float16* __restrict__ wt) {
  int o = blockIdx.x * 256 + threadIdx.x;    // [0, 98304)
  int k = o >> 7, h = o & 127;
  wt[o] = (_Float16)head_w[(size_t)h * IN_DIM + k];
}

// ---------------------------------------------------------------------------
// Stage A: x = embeds @ head_w^T + b, l2-normalize rows; emit nx in BOTH
// row-major (nxh) and per-class transposed (nxhT[c][k][n]) f16 layouts so the
// gram kernel needs no transpose. One block per class (128 rows).
// f16 WMMA 16x16x32, K chunked by 64 (12 chunks); B chunks async-copied.
// ---------------------------------------------------------------------------
__global__ __launch_bounds__(256) void stage_a_kernel(
    const float* __restrict__ embeds, const _Float16* __restrict__ wt,
    const float* __restrict__ head_b, _Float16* __restrict__ nxh,
    _Float16* __restrict__ nxhT) {
  __shared__ __align__(16) unsigned char smem[65536];
  _Float16* Ah = (_Float16*)smem;            // [128][64]
  _Float16* Bt = Ah + 128 * 64;              // [64][128] = wT chunk
  float*    xs = (float*)smem;               // [128][128] aliases Ah/Bt
  __shared__ float invs[128];

  const int t    = threadIdx.x;
  const int lane = t & 31;
  const int w    = t >> 5;
  const int cls  = blockIdx.x;
  const int rb   = cls * 128;                // global row base
  const int wr   = (w >> 1) * 32;            // wave row base in tile
  const int wc   = (w & 1) * 64;             // wave col base in tile

  v8f z;
#pragma unroll
  for (int e = 0; e < 8; ++e) z[e] = 0.0f;
  v8f acc[2][4];
#pragma unroll
  for (int i = 0; i < 2; ++i)
#pragma unroll
    for (int j = 0; j < 4; ++j) acc[i][j] = z;

  for (int kc = 0; kc < IN_DIM / 64; ++kc) {
    // B chunk: straight contiguous copy of wT rows [kc*64, kc*64+64).
#ifdef HAVE_ASYNC_LDS
#pragma unroll
    for (int j = 0; j < 4; ++j) {
      int o = (t + 256 * j) * 8;             // [0, 8192) halves
      async_copy16(wt + (size_t)kc * 8192 + o, Bt + o);
    }
#else
#pragma unroll
    for (int j = 0; j < 4; ++j) {
      int o = (t + 256 * j) * 8;
      *(h8*)(Bt + o) = *(const h8*)(wt + (size_t)kc * 8192 + o);
    }
#endif
    // A chunk: embeds rows, f32 -> f16 (conversion, so regular path).
#pragma unroll
    for (int j = 0; j < 4; ++j) {
      int f = (t + 256 * j) * 8;             // half index in [0, 8192)
      int r = f >> 6, k = f & 63;
      const float* g = embeds + (size_t)(rb + r) * IN_DIM + kc * 64 + k;
      f4 v0 = *(const f4*)g;
      f4 v1 = *(const f4*)(g + 4);
      h8 h;
      h[0] = (_Float16)v0[0]; h[1] = (_Float16)v0[1];
      h[2] = (_Float16)v0[2]; h[3] = (_Float16)v0[3];
      h[4] = (_Float16)v1[0]; h[5] = (_Float16)v1[1];
      h[6] = (_Float16)v1[2]; h[7] = (_Float16)v1[3];
      *(h8*)(Ah + f) = h;
    }
#ifdef HAVE_ASYNC_LDS
    wait_async0();
#endif
    __syncthreads();
#pragma unroll
    for (int k0 = 0; k0 < 64; k0 += 32) {
      v16h af[2], bf[4];
      const int kbA = k0 + (lane >> 4) * 8;  // A layout: lanes>=16 hold K+8
#pragma unroll
      for (int rt = 0; rt < 2; ++rt) {
        int m = wr + rt * 16 + (lane & 15);
        h8 lo = *(const h8*)(Ah + m * 64 + kbA);
        h8 hi = *(const h8*)(Ah + m * 64 + kbA + 16);
        af[rt] = SHUF16(lo, hi);
      }
      const int kbB = k0 + (lane >> 4) * 8;  // B layout: lane == N col, K as A
#pragma unroll
      for (int ct = 0; ct < 4; ++ct) {
        int n = wc + ct * 16 + (lane & 15);
#pragma unroll
        for (int s = 0; s < 8; ++s) {
          bf[ct][s] = Bt[(kbB + s) * 128 + n];
          bf[ct][s + 8] = Bt[(kbB + 16 + s) * 128 + n];
        }
      }
#pragma unroll
      for (int rt = 0; rt < 2; ++rt)
#pragma unroll
        for (int ct = 0; ct < 4; ++ct)
          acc[rt][ct] = __builtin_amdgcn_wmma_f32_16x16x32_f16(
              false, af[rt], false, bf[ct], (short)0, acc[rt][ct], false, false);
    }
    __syncthreads();
  }
  // Add bias in registers, spill x into xs (aliases Ah/Bt; barrier passed).
#pragma unroll
  for (int rt = 0; rt < 2; ++rt)
#pragma unroll
    for (int ct = 0; ct < 4; ++ct) {
      int n = wc + ct * 16 + (lane & 15);
      float bias = head_b[n];
      int mbase = wr + rt * 16 + ((lane >> 4) << 3);
#pragma unroll
      for (int v = 0; v < 8; ++v) {
        acc[rt][ct][v] += bias;
        xs[(mbase + v) * 128 + n] = acc[rt][ct][v];
      }
    }
  __syncthreads();
  // Row l2-norm: thread pair (t, t^1) shares one row via lane shuffle.
  {
    int r = t >> 1, half = (t & 1) * 64;
    const float* row = xs + r * 128 + half;
    float s = 0.0f;
    for (int i = 0; i < 64; ++i) { float x = row[i]; s += x * x; }
    float tot = s + __shfl_xor(s, 1, 32);
    if ((t & 1) == 0) invs[r] = 1.0f / fmaxf(sqrtf(tot), 1e-12f);
  }
  __syncthreads();
  // Row-major output (coalesced).
  for (int e = 0; e < 64; ++e) {
    int flat = t + 256 * e;                  // [0, 16384)
    nxh[(size_t)rb * 128 + flat] = (_Float16)(xs[flat] * invs[flat >> 7]);
  }
  // Transposed output straight from registers: nxhT[cls][k=n][r=mbase+v],
  // 8 contiguous halves per (rt,ct) -> vector stores.
#pragma unroll
  for (int rt = 0; rt < 2; ++rt)
#pragma unroll
    for (int ct = 0; ct < 4; ++ct) {
      int n = wc + ct * 16 + (lane & 15);
      int mbase = wr + rt * 16 + ((lane >> 4) << 3);
      h8 o;
#pragma unroll
      for (int v = 0; v < 8; ++v)
        o[v] = (_Float16)(acc[rt][ct][v] * invs[mbase + v]);
      *(h8*)(nxhT + (size_t)cls * 16384 + n * 128 + mbase) = o;
    }
}

// ---------------------------------------------------------------------------
// Normalize prototypes (f32, tiny): one block per class.
// ---------------------------------------------------------------------------
__global__ __launch_bounds__(128) void proto_norm_kernel(
    const float* __restrict__ proto, float* __restrict__ nproto) {
  __shared__ float red[128];
  int c = blockIdx.x, t = threadIdx.x;
  float v = proto[c * 128 + t];
  red[t] = v * v;
  __syncthreads();
  for (int off = 64; off > 0; off >>= 1) {
    if (t < off) red[t] += red[t + off];
    __syncthreads();
  }
  float inv = 1.0f / fmaxf(sqrtf(red[0]), 1e-12f);
  nproto[c * 128 + t] = v * inv;
}

// ---------------------------------------------------------------------------
// Proto contrastive term: one wave per (c,n) row; lane holds 4 of 128 dims.
// Per-block partial written to pl[blockIdx] (deterministic).
// ---------------------------------------------------------------------------
__global__ __launch_bounds__(256) void proto_loss_kernel(
    const _Float16* __restrict__ nxh, const float* __restrict__ nproto,
    float* __restrict__ pl) {
  __shared__ float wred[8];
  int t = threadIdx.x, lane = t & 31, w = t >> 5;
  int row = blockIdx.x * 8 + w;              // [0, 8192)
  int c = row >> 7;
  h4 xh = *(const h4*)(nxh + (size_t)row * 128 + lane * 4);
  float x0 = (float)xh[0], x1 = (float)xh[1], x2 = (float)xh[2], x3 = (float)xh[3];
  float total = 0.0f, dc = 0.0f;
  for (int k = 0; k < NCLS; ++k) {
    f4 p = *(const f4*)(nproto + k * 128 + lane * 4);
    float d = x0 * p[0] + x1 * p[1] + x2 * p[2] + x3 * p[3];
#pragma unroll
    for (int off = 16; off > 0; off >>= 1) d += __shfl_xor(d, off, 32);
    total += expf(d * INV_SCALE);
    if (k == c) dc = d;
  }
  float part = logf(total) - INV_SCALE * dc; // == -log(pos/total)
  if (lane == 0) wred[w] = part;
  __syncthreads();
  if (t == 0) {
    float s = 0.0f;
    for (int i = 0; i < 8; ++i) s += wred[i];
    pl[blockIdx.x] = s;
  }
}

// Deterministic column-sum reduction over the 64 row-classes.
__global__ __launch_bounds__(256) void colsum_reduce_kernel(
    const float* __restrict__ colPart, float* __restrict__ colSum) {
  int j = blockIdx.x * 256 + threadIdx.x;    // [0, 8192)
  float s = 0.0f;
  for (int aa = 0; aa < NCLS; ++aa) s += colPart[(size_t)aa * 8192 + j];
  colSum[j] = s;
}

// ---------------------------------------------------------------------------
// Instance-instance loss: per class b, per column n:
//   neg = colSum[j] - sum_m pos;  loss += sum_m [log(pos+neg) - log(pos)]
// ---------------------------------------------------------------------------
__global__ __launch_bounds__(128) void inst_loss_kernel(
    const float* __restrict__ diag, const float* __restrict__ colSum,
    float* __restrict__ il) {
  __shared__ float red[128];
  int b = blockIdx.x, n = threadIdx.x;
  const float* blk = diag + (size_t)b * 16384;
  float dsum = 0.0f;
  for (int m = 0; m < 128; ++m) dsum += blk[m * 128 + n];
  float negc = colSum[b * 128 + n] - dsum;
  float s = 0.0f;
  for (int m = 0; m < 128; ++m) {
    float p = blk[m * 128 + n];
    s += logf(p + negc) - logf(p);
  }
  red[n] = s;
  __syncthreads();
  for (int off = 64; off > 0; off >>= 1) {
    if (n < off) red[n] += red[n + off];
    __syncthreads();
  }
  if (n == 0) il[b] = red[0];
}

__global__ void finalize_kernel(const float* __restrict__ pl,
                                const float* __restrict__ il,
                                float* __restrict__ out) {
  float a0 = 0.0f;
  for (int i = 0; i < 1024; ++i) a0 += pl[i];
  float a1 = 0.0f;
  for (int i = 0; i < 64; ++i) a1 += il[i];
  // loss/(num*C*C) + loss_ins/(num*C*num*C)
  out[0] = a0 * (1.0f / 524288.0f) + a1 * (1.0f / 67108864.0f);
}

extern "C" void kernel_launch(void* const* d_in, const int* in_sizes, int n_in,
                              void* d_out, int out_size, void* d_ws, size_t ws_size,
                              hipStream_t stream) {
  const float* embeds = (const float*)d_in[0];   // [64,128,768]
  const float* head_w = (const float*)d_in[1];   // [128,768]
  const float* head_b = (const float*)d_in[2];   // [128]
  const float* proto  = (const float*)d_in[3];   // [64,128]
  float* out = (float*)d_out;

  char* ws = (char*)d_ws;
  _Float16* nxh   = (_Float16*)(ws + OFF_NXH);
  _Float16* nxhT  = (_Float16*)(ws + OFF_NXHT);
  float* nproto   = (float*)(ws + OFF_NPROTO);
  float* colPart  = (float*)(ws + OFF_COLPART);
  float* colSum   = (float*)(ws + OFF_COLSUM);
  float* diag     = (float*)(ws + OFF_DIAG);
  float* pl       = (float*)(ws + OFF_PL);
  float* il       = (float*)(ws + OFF_IL);
  _Float16* wt    = (_Float16*)(ws + OFF_WT);

  prep_wt_kernel<<<384, 256, 0, stream>>>(head_w, wt);
  stage_a_kernel<<<64, 256, 0, stream>>>(embeds, wt, head_b, nxh, nxhT);
  proto_norm_kernel<<<64, 128, 0, stream>>>(proto, nproto);
  proto_loss_kernel<<<1024, 256, 0, stream>>>(nxh, nproto, pl);
  gram_kernel<<<dim3(64, 64), 256, 0, stream>>>(nxh, nxhT, colPart, diag);
  colsum_reduce_kernel<<<32, 256, 0, stream>>>(colPart, colSum);
  inst_loss_kernel<<<64, 128, 0, stream>>>(diag, colSum, il);
  finalize_kernel<<<1, 1, 0, stream>>>(pl, il, out);
}
